// MoE_9526237463019
// MI455X (gfx1250) — matrix operand order repaired and checked by reading, not compiled, not run on hardware
//
#include <hip/hip_runtime.h>
#include <hip/hip_bf16.h>

// ---------------- problem constants ----------------
#define TT   4096      // B*S tokens
#define HH   768
#define FF   3072
#define EE   8
#define NSLOT (TT*2)   // total expert assignments (top-2, always exactly T*K)

#define TILE 128
#define KCH  32
#define LDS_STRIDE 40  // bf16 elements per LDS row (32 data + 8 pad = 80B -> conflict-free)

typedef __bf16 bf16_t;
typedef __attribute__((ext_vector_type(16))) __bf16 v16bf;
typedef __attribute__((ext_vector_type(8)))  float  v8f;
typedef __attribute__((ext_vector_type(4))) unsigned int u32x4;
typedef __attribute__((ext_vector_type(8))) int          i32x8;
typedef __attribute__((ext_vector_type(4))) int          i32x4;
typedef int v2i_vs __attribute__((vector_size(8)));   // matches builtin's V2i pointee

#define GASP __attribute__((address_space(1)))
#define LASP __attribute__((address_space(3)))

// ------- CDNA5 feature gates (guarded: plain-copy fallback keeps compile green) -------
#if defined(__HIP_DEVICE_COMPILE__)
#if __has_builtin(__builtin_amdgcn_global_load_async_to_lds_b64) && __has_builtin(__builtin_amdgcn_s_wait_asynccnt)
#define USE_ASYNC 1
#endif
#if __has_builtin(__builtin_amdgcn_tensor_load_to_lds) && __has_builtin(__builtin_amdgcn_s_wait_tensorcnt)
#define USE_TDM 1
#endif
#endif
#ifndef USE_ASYNC
#define USE_ASYNC 0
#endif
#ifndef USE_TDM
#define USE_TDM 0
#endif

union FragU { v16bf v; unsigned int u[8]; };

// round-to-nearest-even fp32 -> bf16
__device__ __forceinline__ bf16_t f2bf(float x) {
  unsigned int u = __float_as_uint(x);
  u += 0x7FFFu + ((u >> 16) & 1u);
  unsigned short s = (unsigned short)(u >> 16);
  bf16_t r; __builtin_memcpy(&r, &s, 2); return r;
}
__device__ __forceinline__ unsigned int pack_bf16x2(float a, float b) {
  unsigned int ua = __float_as_uint(a);
  unsigned int ub = __float_as_uint(b);
  ua += 0x7FFFu + ((ua >> 16) & 1u);
  ub += 0x7FFFu + ((ub >> 16) & 1u);
  return (ua >> 16) | (ub & 0xFFFF0000u);
}

// 8-byte global -> LDS copy: async DMA on CDNA5, VGPR bounce otherwise.
__device__ __forceinline__ void copy8(const bf16_t* g, bf16_t* l) {
#if USE_ASYNC
  __builtin_amdgcn_global_load_async_to_lds_b64(
      (GASP v2i_vs*)(size_t)(g),
      (LASP v2i_vs*)(unsigned int)(size_t)(l), 0, 0);
#else
  *(uint2*)l = *(const uint2*)g;
#endif
}
__device__ __forceinline__ void async_join() {
#if USE_ASYNC
  __builtin_amdgcn_s_wait_asynccnt(0);
#endif
}

// Tensor Data Mover: 2D tile (128 rows x 32 bf16) from row-major [rows][kdim] bf16
// into LDS with 16B padding after every 64B row -> matches LDS_STRIDE=40 layout.
__device__ __forceinline__ void tdm_load_tile(const bf16_t* gsrc, bf16_t* lds, int kdim) {
#if USE_TDM
  unsigned long long ga = (unsigned long long)(size_t)gsrc;
  u32x4 g0;
  g0[0] = 1u;                                              // count=1, user mode
  g0[1] = (unsigned int)(size_t)lds;                       // lds byte address
  g0[2] = (unsigned int)ga;                                // global addr lo
  g0[3] = (unsigned int)((ga >> 32) & 0x1FFFFFFu) | (2u << 30);  // addr hi | type=2
  i32x8 g1;
  g1[0] = (1 << 16) | (1 << 20) | (3 << 22) | (3 << 25);   // 2B elems, pad_en, 16DW int, 4DW pad
  g1[1] = (kdim & 0xFFFF) << 16;                           // tensor_dim0[15:0]
  g1[2] = ((unsigned)kdim >> 16) | (4096 << 16);           // dim0 hi | tensor_dim1 lo
  g1[3] = (32 << 16);                                      // dim1 hi=0 | tile_dim0=32
  g1[4] = 128;                                             // tile_dim1=128, tile_dim2=0
  g1[5] = kdim;                                            // tensor_dim0_stride lo32 (elems)
  g1[6] = 0; g1[7] = 0;
  i32x4 z4 = {};
#if __clang_major__ >= 23
  i32x8 z8 = {};
  __builtin_amdgcn_tensor_load_to_lds(g0, g1, z4, z4, z8, 0);
#else
  __builtin_amdgcn_tensor_load_to_lds(g0, g1, z4, z4, 0);
#endif
#endif
}
__device__ __forceinline__ void tdm_join() {
#if USE_TDM
  __builtin_amdgcn_s_wait_tensorcnt(0);
#endif
}

// A-fragment (16x32 bf16), ISA 7.12.2: lanes 0-15 M=lane, K={0..7,16..23};
// lanes 16-31 M=lane-16, K={8..15,24..31}.
__device__ __forceinline__ v16bf load_fragA(const bf16_t* tile, int rowBase, int lane) {
  int r = rowBase + (lane & 15);
  int kb = (lane & 16) ? 4 : 0;
  const unsigned int* p = (const unsigned int*)(tile + r * LDS_STRIDE);
  FragU f;
  f.u[0] = p[kb + 0]; f.u[1] = p[kb + 1]; f.u[2] = p[kb + 2]; f.u[3] = p[kb + 3];
  f.u[4] = p[kb + 8]; f.u[5] = p[kb + 9]; f.u[6] = p[kb + 10]; f.u[7] = p[kb + 11];
  return f.v;
}
// B-fragment (32x16 bf16): lane half -> K {0..7,16..23} / {8..15,24..31}; tile stored N-major [n][k].
__device__ __forceinline__ v16bf load_fragB(const bf16_t* tile, int colBase, int lane) {
  int n = colBase + (lane & 15);
  int kb = (lane & 16) ? 4 : 0;
  const unsigned int* p = (const unsigned int*)(tile + n * LDS_STRIDE);
  FragU f;
#pragma unroll
  for (int i = 0; i < 8; ++i) f.u[i] = p[kb + i + (i >= 4 ? 4 : 0)];
  return f.v;
}

// ---------------- fp32 -> bf16 convert (contiguous) ----------------
__global__ __launch_bounds__(256)
void cvt_kernel(const float* __restrict__ src, bf16_t* __restrict__ dst, int n4) {
  int i = blockIdx.x * 256 + threadIdx.x;
  if (i >= n4) return;
  float4 v = ((const float4*)src)[i];
  ((uint2*)dst)[i] = make_uint2(pack_bf16x2(v.x, v.y), pack_bf16x2(v.z, v.w));
}

// ---------------- fp32 [R][C] -> bf16 [C][R] transpose-convert (per blockIdx.z matrix) ----
__global__ __launch_bounds__(256)
void transpose_cvt_kernel(const float* __restrict__ src, bf16_t* __restrict__ dst,
                          int R, int C) {
  __shared__ float tile[32][33];
  const size_t zoff = (size_t)blockIdx.z * R * C;
  const int c0 = blockIdx.x * 32, r0 = blockIdx.y * 32;
  const int tx = threadIdx.x & 31, ty = threadIdx.x >> 5;   // ty: 0..7
#pragma unroll
  for (int q = 0; q < 4; ++q)
    tile[ty + q * 8][tx] = src[zoff + (size_t)(r0 + ty + q * 8) * C + c0 + tx];
  __syncthreads();
#pragma unroll
  for (int q = 0; q < 4; ++q)
    dst[zoff + (size_t)(c0 + ty + q * 8) * R + r0 + tx] = f2bf(tile[tx][ty + q * 8]);
}

// ---------------- router: logits, top-2, softmax, counts ----------------
__global__ __launch_bounds__(256)
void router_kernel(const float* __restrict__ flat, const float* __restrict__ Wr,
                   const float* __restrict__ br, int* __restrict__ cnt,
                   int* __restrict__ tok_e, float* __restrict__ tok_g) {
  int t = blockIdx.x * 256 + threadIdx.x;
  if (t >= TT) return;
  float acc[EE];
#pragma unroll
  for (int e = 0; e < EE; ++e) acc[e] = br[e];
  const float* x = flat + (size_t)t * HH;
  for (int h = 0; h < HH; ++h) {
    float xv = x[h];
#pragma unroll
    for (int e = 0; e < EE; ++e) acc[e] = fmaf(xv, Wr[e * HH + h], acc[e]);
  }
  int e0 = 0; float v0 = acc[0];
#pragma unroll
  for (int e = 1; e < EE; ++e) if (acc[e] > v0) { v0 = acc[e]; e0 = e; }
  int e1 = (e0 == 0) ? 1 : 0; float v1 = acc[e1];
#pragma unroll
  for (int e = 0; e < EE; ++e) if (e != e0 && acc[e] > v1) { v1 = acc[e]; e1 = e; }
  float ex = __expf(v1 - v0);
  float z  = 1.0f + ex;
  tok_e[t * 2 + 0] = e0; tok_g[t * 2 + 0] = 1.0f / z;
  tok_e[t * 2 + 1] = e1; tok_g[t * 2 + 1] = ex / z;
  atomicAdd(&cnt[e0], 1);
  atomicAdd(&cnt[e1], 1);
}

__global__ void prefix_kernel(const int* __restrict__ cnt, int* __restrict__ basev,
                              int* __restrict__ cursor) {
  if (threadIdx.x == 0) {
    int s = 0;
    for (int e = 0; e < EE; ++e) { basev[e] = s; s += cnt[e]; cursor[e] = 0; }
  }
}

__global__ __launch_bounds__(256)
void scatter_kernel(const int* __restrict__ tok_e, const int* __restrict__ basev,
                    int* __restrict__ cursor, int* __restrict__ assign_tok,
                    int* __restrict__ slot_of) {
  int t = blockIdx.x * 256 + threadIdx.x;
  if (t >= TT) return;
#pragma unroll
  for (int k = 0; k < 2; ++k) {
    int e = tok_e[t * 2 + k];
    int pos = atomicAdd(&cursor[e], 1);
    int slot = basev[e] + pos;
    assign_tok[slot] = t;
    slot_of[t * 2 + k] = slot;
  }
}

// ---------------- GEMM1: h1 = gelu(gather(xbuf) @ W1T[e]^T + b1[e]) ----------------
__global__ __launch_bounds__(256)
void gemm1_kernel(const bf16_t* __restrict__ xbuf, const bf16_t* __restrict__ W1T,
                  const float* __restrict__ b1, const int* __restrict__ cnt,
                  const int* __restrict__ basev, const int* __restrict__ assign_tok,
                  bf16_t* __restrict__ h1buf) {
  __shared__ bf16_t At[2][TILE * LDS_STRIDE];
  __shared__ bf16_t Bt[2][TILE * LDS_STRIDE];
  const int e  = blockIdx.z;
  const int ce = cnt[e];
  const int mb = blockIdx.y * TILE;
  if (mb >= ce) return;
  const int nb = blockIdx.x * TILE;
  const int be = basev[e];
  const int tid = threadIdx.x;
  const int c4 = tid & 7;
  const int wave = tid >> 5, lane = tid & 31;
  const int wm = (wave >> 1) * 32;
  const int wn = (wave & 1) * 64;

  const bf16_t* as_[4];
  const bf16_t* bs_[4];
  int idx_[4];
#pragma unroll
  for (int j = 0; j < 4; ++j) {
    int r = (tid >> 3) + j * 32;
    int row = mb + r; if (row >= ce) row = ce - 1;
    as_[j] = xbuf + (size_t)assign_tok[be + row] * HH + c4 * 4;
    bs_[j] = W1T + ((size_t)e * FF + nb + r) * HH + c4 * 4;
    idx_[j] = r * LDS_STRIDE + c4 * 4;
  }
  const bf16_t* btile = W1T + ((size_t)e * FF + nb) * HH;

  auto stage = [&](int b, int k0) {
#pragma unroll
    for (int j = 0; j < 4; ++j) copy8(as_[j] + k0, &At[b][idx_[j]]);
#if USE_TDM
    if (wave == 0) tdm_load_tile(btile + k0, &Bt[b][0], HH);
#else
#pragma unroll
    for (int j = 0; j < 4; ++j) copy8(bs_[j] + k0, &Bt[b][idx_[j]]);
#endif
  };

  v8f acc[2][4] = {};
  stage(0, 0);
  async_join();
  if (wave == 0) tdm_join();
  __syncthreads();

  int buf = 0;
  for (int k0 = 0; k0 < HH; k0 += KCH) {
    int nk = k0 + KCH;
    if (nk < HH) stage(buf ^ 1, nk);
    if (k0 + 2 * KCH < HH) __builtin_prefetch((const void*)(as_[0] + k0 + 2 * KCH), 0, 0);

    v16bf a0 = load_fragA(&At[buf][0], wm,      lane);
    v16bf a1 = load_fragA(&At[buf][0], wm + 16, lane);
    v16bf b0 = load_fragB(&Bt[buf][0], wn,      lane);
    v16bf b1f = load_fragB(&Bt[buf][0], wn + 16, lane);
    v16bf b2f = load_fragB(&Bt[buf][0], wn + 32, lane);
    v16bf b3f = load_fragB(&Bt[buf][0], wn + 48, lane);
    acc[0][0] = __builtin_amdgcn_wmma_f32_16x16x32_bf16(false, a0, false, b0,  (short)0, acc[0][0], false, false);
    acc[0][1] = __builtin_amdgcn_wmma_f32_16x16x32_bf16(false, a0, false, b1f, (short)0, acc[0][1], false, false);
    acc[0][2] = __builtin_amdgcn_wmma_f32_16x16x32_bf16(false, a0, false, b2f, (short)0, acc[0][2], false, false);
    acc[0][3] = __builtin_amdgcn_wmma_f32_16x16x32_bf16(false, a0, false, b3f, (short)0, acc[0][3], false, false);
    acc[1][0] = __builtin_amdgcn_wmma_f32_16x16x32_bf16(false, a1, false, b0,  (short)0, acc[1][0], false, false);
    acc[1][1] = __builtin_amdgcn_wmma_f32_16x16x32_bf16(false, a1, false, b1f, (short)0, acc[1][1], false, false);
    acc[1][2] = __builtin_amdgcn_wmma_f32_16x16x32_bf16(false, a1, false, b2f, (short)0, acc[1][2], false, false);
    acc[1][3] = __builtin_amdgcn_wmma_f32_16x16x32_bf16(false, a1, false, b3f, (short)0, acc[1][3], false, false);

    async_join();
    if (wave == 0) tdm_join();
    __syncthreads();
    buf ^= 1;
  }

  const int half = (lane >> 4) & 1;
  const int nl = lane & 15;
#pragma unroll
  for (int i = 0; i < 2; ++i)
#pragma unroll
    for (int j = 0; j < 4; ++j)
#pragma unroll
      for (int g = 0; g < 8; ++g) {
        int m = wm + i * 16 + g + half * 8;
        int row = mb + m;
        if (row < ce) {
          int n = nb + wn + j * 16 + nl;
          float xv = acc[i][j][g] + b1[e * FF + n];
          float ge = 0.5f * xv * (1.0f + erff(xv * 0.70710678118654752f));
          h1buf[(size_t)(be + row) * FF + n] = f2bf(ge);
        }
      }
}

// ---------------- GEMM2: y = h1 @ W2T[e]^T + b2[e] ----------------
__global__ __launch_bounds__(256)
void gemm2_kernel(const bf16_t* __restrict__ h1buf, const bf16_t* __restrict__ W2T,
                  const float* __restrict__ b2, const int* __restrict__ cnt,
                  const int* __restrict__ basev, float* __restrict__ ybuf) {
  __shared__ bf16_t At[2][TILE * LDS_STRIDE];
  __shared__ bf16_t Bt[2][TILE * LDS_STRIDE];
  const int e  = blockIdx.z;
  const int ce = cnt[e];
  const int mb = blockIdx.y * TILE;
  if (mb >= ce) return;
  const int nb = blockIdx.x * TILE;
  const int be = basev[e];
  const int tid = threadIdx.x;
  const int c4 = tid & 7;
  const int wave = tid >> 5, lane = tid & 31;
  const int wm = (wave >> 1) * 32;
  const int wn = (wave & 1) * 64;

  const bf16_t* as_[4];
  const bf16_t* bs_[4];
  int idx_[4];
#pragma unroll
  for (int j = 0; j < 4; ++j) {
    int r = (tid >> 3) + j * 32;
    int row = mb + r; if (row >= ce) row = ce - 1;
    as_[j] = h1buf + (size_t)(be + row) * FF + c4 * 4;
    bs_[j] = W2T + ((size_t)e * HH + nb + r) * FF + c4 * 4;
    idx_[j] = r * LDS_STRIDE + c4 * 4;
  }
  const bf16_t* btile = W2T + ((size_t)e * HH + nb) * FF;

  auto stage = [&](int b, int k0) {
#pragma unroll
    for (int j = 0; j < 4; ++j) copy8(as_[j] + k0, &At[b][idx_[j]]);
#if USE_TDM
    if (wave == 0) tdm_load_tile(btile + k0, &Bt[b][0], FF);
#else
#pragma unroll
    for (int j = 0; j < 4; ++j) copy8(bs_[j] + k0, &Bt[b][idx_[j]]);
#endif
  };

  v8f acc[2][4] = {};
  stage(0, 0);
  async_join();
  if (wave == 0) tdm_join();
  __syncthreads();

  int buf = 0;
  for (int k0 = 0; k0 < FF; k0 += KCH) {
    int nk = k0 + KCH;
    if (nk < FF) stage(buf ^ 1, nk);
    if (k0 + 2 * KCH < FF) __builtin_prefetch((const void*)(as_[0] + k0 + 2 * KCH), 0, 0);

    v16bf a0 = load_fragA(&At[buf][0], wm,      lane);
    v16bf a1 = load_fragA(&At[buf][0], wm + 16, lane);
    v16bf b0 = load_fragB(&Bt[buf][0], wn,      lane);
    v16bf b1f = load_fragB(&Bt[buf][0], wn + 16, lane);
    v16bf b2f = load_fragB(&Bt[buf][0], wn + 32, lane);
    v16bf b3f = load_fragB(&Bt[buf][0], wn + 48, lane);
    acc[0][0] = __builtin_amdgcn_wmma_f32_16x16x32_bf16(false, a0, false, b0,  (short)0, acc[0][0], false, false);
    acc[0][1] = __builtin_amdgcn_wmma_f32_16x16x32_bf16(false, a0, false, b1f, (short)0, acc[0][1], false, false);
    acc[0][2] = __builtin_amdgcn_wmma_f32_16x16x32_bf16(false, a0, false, b2f, (short)0, acc[0][2], false, false);
    acc[0][3] = __builtin_amdgcn_wmma_f32_16x16x32_bf16(false, a0, false, b3f, (short)0, acc[0][3], false, false);
    acc[1][0] = __builtin_amdgcn_wmma_f32_16x16x32_bf16(false, a1, false, b0,  (short)0, acc[1][0], false, false);
    acc[1][1] = __builtin_amdgcn_wmma_f32_16x16x32_bf16(false, a1, false, b1f, (short)0, acc[1][1], false, false);
    acc[1][2] = __builtin_amdgcn_wmma_f32_16x16x32_bf16(false, a1, false, b2f, (short)0, acc[1][2], false, false);
    acc[1][3] = __builtin_amdgcn_wmma_f32_16x16x32_bf16(false, a1, false, b3f, (short)0, acc[1][3], false, false);

    async_join();
    if (wave == 0) tdm_join();
    __syncthreads();
    buf ^= 1;
  }

  const int half = (lane >> 4) & 1;
  const int nl = lane & 15;
#pragma unroll
  for (int i = 0; i < 2; ++i)
#pragma unroll
    for (int j = 0; j < 4; ++j)
#pragma unroll
      for (int g = 0; g < 8; ++g) {
        int m = wm + i * 16 + g + half * 8;
        int row = mb + m;
        if (row < ce) {
          int n = nb + wn + j * 16 + nl;
          ybuf[(size_t)(be + row) * HH + n] = acc[i][j][g] + b2[e * HH + n];
        }
      }
}

// ---------------- residual + LayerNorm combine ----------------
__global__ __launch_bounds__(256)
void ln_kernel(const float* __restrict__ flat, const float* __restrict__ ybuf,
               const int* __restrict__ slot_of, const float* __restrict__ tok_g,
               const float* __restrict__ lnw, const float* __restrict__ lnb,
               float* __restrict__ out) {
  __shared__ float red[16];
  const int t = blockIdx.x;
  const int tid = threadIdx.x;
  const int wave = tid >> 5, lane = tid & 31;
  const size_t s0 = (size_t)slot_of[t * 2 + 0] * HH;
  const size_t s1 = (size_t)slot_of[t * 2 + 1] * HH;
  const float g0 = tok_g[t * 2 + 0], g1 = tok_g[t * 2 + 1];

  float v[3]; float sum = 0.f;
#pragma unroll
  for (int q = 0; q < 3; ++q) {
    int h = tid + q * 256;
    float x = flat[(size_t)t * HH + h] + g0 * ybuf[s0 + h] + g1 * ybuf[s1 + h];
    v[q] = x; sum += x;
  }
  for (int o = 16; o > 0; o >>= 1) sum += __shfl_down(sum, o, 32);
  if (lane == 0) red[wave] = sum;
  __syncthreads();
  if (tid == 0) { float s = 0.f; for (int w = 0; w < 8; ++w) s += red[w]; red[8] = s; }
  __syncthreads();
  const float mu = red[8] * (1.0f / HH);

  float sq = 0.f;
#pragma unroll
  for (int q = 0; q < 3; ++q) { float d = v[q] - mu; sq += d * d; }
  for (int o = 16; o > 0; o >>= 1) sq += __shfl_down(sq, o, 32);
  __syncthreads();
  if (lane == 0) red[wave] = sq;
  __syncthreads();
  if (tid == 0) { float s = 0.f; for (int w = 0; w < 8; ++w) s += red[w]; red[9] = s; }
  __syncthreads();
  const float var = red[9] * (1.0f / HH);
  const float inv = rsqrtf(var + 1e-12f);
#pragma unroll
  for (int q = 0; q < 3; ++q) {
    int h = tid + q * 256;
    out[(size_t)t * HH + h] = (v[q] - mu) * inv * lnw[h] + lnb[h];
  }
}

// ---------------- launch ----------------
extern "C" void kernel_launch(void* const* d_in, const int* in_sizes, int n_in,
                              void* d_out, int out_size, void* d_ws, size_t ws_size,
                              hipStream_t stream) {
  const float* flat = (const float*)d_in[0];
  const float* Wr   = (const float*)d_in[1];
  const float* br   = (const float*)d_in[2];
  const float* W1   = (const float*)d_in[3];
  const float* b1   = (const float*)d_in[4];
  const float* W2   = (const float*)d_in[5];
  const float* b2   = (const float*)d_in[6];
  const float* lnw  = (const float*)d_in[7];
  const float* lnb  = (const float*)d_in[8];
  float* out = (float*)d_out;

  char* ws = (char*)d_ws;
  int*    cnt        = (int*)   (ws + 0);
  int*    cursor     = (int*)   (ws + 64);
  int*    basev      = (int*)   (ws + 128);
  int*    tok_e      = (int*)   (ws + 256);
  float*  tok_g      = (float*) (ws + 33024);
  int*    assign_tok = (int*)   (ws + 65792);
  int*    slot_of    = (int*)   (ws + 98560);
  bf16_t* xbuf       = (bf16_t*)(ws + 131328);     // T*H bf16      = 6 MB
  bf16_t* W1T        = (bf16_t*)(ws + 6422784);    // E*F*H bf16    = 36 MB  (transposed)
  bf16_t* W2T        = (bf16_t*)(ws + 44171520);   // E*H*F bf16    = 36 MB  (transposed)
  bf16_t* h1buf      = (bf16_t*)(ws + 81920256);   // NSLOT*F bf16  = 48 MB
  float*  ybuf       = (float*) (ws + 132251904);  // NSLOT*H f32   = 24 MB

  (void)hipMemsetAsync(cnt, 0, EE * sizeof(int), stream);

  // one-shot precision/layout conversion (bf16 weights fit in 192MB L2)
  cvt_kernel<<<(TT * HH / 4 + 255) / 256, 256, 0, stream>>>(flat, xbuf, TT * HH / 4);
  transpose_cvt_kernel<<<dim3(FF / 32, HH / 32, EE), 256, 0, stream>>>(W1, W1T, HH, FF);
  transpose_cvt_kernel<<<dim3(HH / 32, FF / 32, EE), 256, 0, stream>>>(W2, W2T, FF, HH);

  router_kernel <<<TT / 256, 256, 0, stream>>>(flat, Wr, br, cnt, tok_e, tok_g);
  prefix_kernel <<<1, 32, 0, stream>>>(cnt, basev, cursor);
  scatter_kernel<<<TT / 256, 256, 0, stream>>>(tok_e, basev, cursor, assign_tok, slot_of);

  gemm1_kernel<<<dim3(FF / TILE, TT / TILE, EE), 256, 0, stream>>>(xbuf, W1T, b1, cnt, basev, assign_tok, h1buf);
  gemm2_kernel<<<dim3(HH / TILE, TT / TILE, EE), 256, 0, stream>>>(h1buf, W2T, b2, cnt, basev, ybuf);
  ln_kernel   <<<TT, 256, 0, stream>>>(flat, ybuf, slot_of, tok_g, lnw, lnb, out);
}
